// MultiHeadAttention_39676907886201
// MI455X (gfx1250) — matrix operand order repaired and checked by reading, not compiled, not run on hardware
//
#include <hip/hip_runtime.h>
#include <hip/hip_bf16.h>

// ---------------------------------------------------------------------------
// MultiHeadAttention for MI455X (gfx1250, wave32, WMMA + TDM).
//
//   1) f32 -> bf16 conversions (A operands) and transposed conversions (weights)
//   2) q = query @ Wq        : 8192x4096x4096 bf16 WMMA GEMM  (~275 GFLOP, dominant)
//      k,v projections       : 8192x4096x64
//   3) per-token 64x64 logits + log-softmax + dropout + @v  (VALU, tiny FLOPs)
//   4) out = attn_flat @ Wo + bo : 8192x64x4096 WMMA GEMM -> f32 d_out
//
// Compute-bound (~830 FLOP/byte). GEMM uses Tensor Data Mover double-buffered
// LDS staging: wave0/wave1 issue 2D-tile tensor_load_to_lds (A: 64x64 bf16,
// B: 256x64 bf16) with TDM padding (32 dwords interval, 4 dwords pad) so LDS
// rows have a 72-element stride -> conflict-free ds_load_b128 fragment reads.
// Sync: s_wait_tensorcnt + workgroup barrier; compute overlaps next-stage DMA.
// ---------------------------------------------------------------------------

typedef __attribute__((ext_vector_type(16))) __bf16 v16bf;
typedef __attribute__((ext_vector_type(8)))  __bf16 v8bf;
typedef __attribute__((ext_vector_type(4)))  __bf16 v4bf;
typedef __attribute__((ext_vector_type(8)))  float  v8f;
typedef __attribute__((ext_vector_type(4)))  unsigned int u32x4;
typedef __attribute__((ext_vector_type(8)))  int    i32x8;
typedef __attribute__((ext_vector_type(4)))  int    i32x4;

#define S_TOK 8192
#define QC    4096
#define HC    4096   // H * KC
#define NH    64

// ---------------- f32 -> bf16 elementwise (x4 vectorized) ----------------
__global__ void cvt_f32_bf16_v4(const float4* __restrict__ in, v4bf* __restrict__ out, long n4) {
    long i = (long)blockIdx.x * blockDim.x + threadIdx.x;
    if (i < n4) {
        float4 f = in[i];
        v4bf o;
        o[0] = (__bf16)f.x; o[1] = (__bf16)f.y; o[2] = (__bf16)f.z; o[3] = (__bf16)f.w;
        out[i] = o;
    }
}

// ---------------- f32 [K][N] -> bf16 [N][K] (transpose) ----------------
__global__ void cvt_transpose_f32_bf16(const float* __restrict__ in, __bf16* __restrict__ out,
                                       int K, int N) {
    long i = (long)blockIdx.x * blockDim.x + threadIdx.x;
    long total = (long)K * N;
    if (i < total) {
        int k = (int)(i / N);
        int n = (int)(i % N);
        out[(long)n * K + k] = (__bf16)in[i];
    }
}

// ---------------- TDM 2D tile load: global -> LDS ----------------
// Loads a [rows x 64] bf16 tile (row stride = row_stride_elems in global) to
// lds_off, inserting 16B of LDS padding after every 128B row (pad_interval=32
// dwords -> code 4; pad_amount=4 dwords -> code 3)  => LDS row stride 144B.
// Descriptor packing per CDNA5 ISA sec 8.3/8.4 (groups 0/1; 2D so groups 2/3
// and the trailing group slot are zero).  This toolchain's builtin arity is 6:
//   (u32x4 g0, i32x8 g1, i32x4, i32x4, i32x8, i32 cpol)
__device__ __forceinline__ void tdm_load_tile_2d(unsigned lds_off, const __bf16* gptr,
                                                 int rows, int tensor_rows,
                                                 long row_stride_elems) {
    unsigned long long ga = (unsigned long long)(const void*)gptr;
    unsigned dim0 = 0x7FFFFFFFu;                     // dim0 never OOB (K mult of 64)
    unsigned dim1 = (unsigned)(tensor_rows > 0 ? tensor_rows : 0);
    unsigned long long s0 = (unsigned long long)row_stride_elems;

    u32x4 g0;
    g0[0] = 1u;                                      // count=1, user mode
    g0[1] = lds_off;                                 // lds_addr
    g0[2] = (unsigned)(ga & 0xFFFFFFFFu);            // global_addr[31:0]
    g0[3] = (unsigned)((ga >> 32) & 0x01FFFFFFu)     // global_addr[56:32]
          | (2u << 30);                              // type=2 ("image")

    i32x8 g1;
    g1[0] = (int)((1u << 16)                         // data_size = 2 bytes
          |       (1u << 20)                         // pad_enable
          |       (4u << 22)                         // pad_interval: 32 dwords
          |       (3u << 25));                       // pad_amount: 4 dwords
    g1[1] = (int)((dim0 & 0xFFFFu) << 16);           // tensor_dim0[15:0]
    g1[2] = (int)((dim0 >> 16) | ((dim1 & 0xFFFFu) << 16));   // dim0 hi | dim1 lo
    g1[3] = (int)((dim1 >> 16) | (64u << 16));       // dim1 hi | tile_dim0 = 64
    g1[4] = (int)((unsigned)rows & 0xFFFFu);         // tile_dim1 = rows, tile_dim2 = 0
    g1[5] = (int)(unsigned)(s0 & 0xFFFFFFFFu);       // tensor_dim0_stride lo
    g1[6] = (int)(unsigned)((s0 >> 32) & 0xFFFFu);   // tensor_dim0_stride hi
    g1[7] = 0;
    i32x4 gz4 = {0, 0, 0, 0};
    i32x8 gz8 = {0, 0, 0, 0, 0, 0, 0, 0};
    __builtin_amdgcn_tensor_load_to_lds(g0, g1, gz4, gz4, gz8, 0);
}

// ---------------- bf16 WMMA GEMM (TDM double-buffered LDS staging) --------
// C[M][N] = A[M][K] @ Bt[N][K]^T + bias[N]
// Block: 256 threads = 8 waves (2 x 4). Block tile 64(M) x 256(N), BK = 64.
// Wave tile: 32 x 64  ->  8 x v_wmma_f32_16x16x32_bf16 per K=32 step.
//
// Fragment layouts per CDNA5 ISA 7.12.2 (wave32):
//  A 16x32 bf16: lane l -> row m = l&15; half = l>>4;
//     elems 0..7  = A[m][k0..k0+7], elems 8..15 = A[m][k0+16..k0+23], k0=ks+half*8
//  B 32x16 bf16: lane l -> col n = l&15; elems 0..7 = Bt[n][k0..k0+7], 8..15 = Bt[n][k0+16..k0+23], k0 as for A
//  C/D 16x16 f32: lane l -> col n = l&15; vgpr r -> row m = r + 8*half
#define LDS_RS 72   // LDS row stride in bf16 elems (64 data + 8 TDM pad)

template <bool OUT_BF16>
__global__ __launch_bounds__(256) void gemm_bf16_wmma_tdm(
        const __bf16* __restrict__ A,
        const __bf16* __restrict__ Bt,
        const float*  __restrict__ bias,
        void* __restrict__ C,
        int M, int N, int K) {
    __shared__ __bf16 Alds[2][64  * LDS_RS];
    __shared__ __bf16 Blds[2][256 * LDS_RS];

    const int lane  = threadIdx.x & 31;
    const int wave  = threadIdx.x >> 5;
    const int tileM = blockIdx.y * 64;
    const int tileN = blockIdx.x * 256;
    const int mwarp = (wave >> 2) * 32;          // wave row offset in block tile
    const int nwarp = (wave & 3) * 64;           // wave col offset in block tile
    const int lm    = lane & 15;
    const int half  = lane >> 4;

    v8f acc[2][4];
#pragma unroll
    for (int i = 0; i < 2; ++i)
#pragma unroll
        for (int j = 0; j < 4; ++j)
#pragma unroll
            for (int r = 0; r < 8; ++r) acc[i][j][r] = 0.0f;

    // ---- initial TDM stage into buffer 0 ----
    if (wave == 0)
        tdm_load_tile_2d((unsigned)(size_t)&Alds[0][0], A + (size_t)tileM * K,
                         64, M - tileM, K);
    if (wave == 1)
        tdm_load_tile_2d((unsigned)(size_t)&Blds[0][0], Bt + (size_t)tileN * K,
                         256, N - tileN, K);
    if (wave < 2) __builtin_amdgcn_s_wait_tensorcnt(0);
    __syncthreads();

    for (int kk = 0; kk < K; kk += 64) {
        const int  buf      = (kk >> 6) & 1;
        const bool has_next = (kk + 64) < K;

        // prefetch next BK window into the other buffer while we compute
        if (has_next) {
            if (wave == 0)
                tdm_load_tile_2d((unsigned)(size_t)&Alds[buf ^ 1][0],
                                 A + (size_t)tileM * K + (kk + 64), 64, M - tileM, K);
            if (wave == 1)
                tdm_load_tile_2d((unsigned)(size_t)&Blds[buf ^ 1][0],
                                 Bt + (size_t)tileN * K + (kk + 64), 256, N - tileN, K);
        }

        const __bf16* As = &Alds[buf][0];
        const __bf16* Bs = &Blds[buf][0];
#pragma unroll
        for (int ks = 0; ks < 64; ks += 32) {
            v16bf af[2], bfr[4];
#pragma unroll
            for (int i = 0; i < 2; ++i) {
                const __bf16* p = As + (mwarp + i * 16 + lm) * LDS_RS + ks + half * 8;
                v8bf lo = *(const v8bf*)p;
                v8bf hi = *(const v8bf*)(p + 16);
                af[i] = __builtin_shufflevector(lo, hi, 0, 1, 2, 3, 4, 5, 6, 7,
                                                8, 9, 10, 11, 12, 13, 14, 15);
            }
#pragma unroll
            for (int j = 0; j < 4; ++j) {
                const __bf16* p = Bs + (nwarp + j * 16 + lm) * LDS_RS + ks + half * 8;
                v8bf lo = *(const v8bf*)p;
                v8bf hi = *(const v8bf*)(p + 16);
                bfr[j] = __builtin_shufflevector(lo, hi, 0, 1, 2, 3, 4, 5, 6, 7,
                                                 8, 9, 10, 11, 12, 13, 14, 15);
            }
#pragma unroll
            for (int i = 0; i < 2; ++i)
#pragma unroll
                for (int j = 0; j < 4; ++j)
                    acc[i][j] = __builtin_amdgcn_wmma_f32_16x16x32_bf16(
                        false, af[i], false, bfr[j], (short)0, acc[i][j], false, false);
        }

        if (has_next && wave < 2) __builtin_amdgcn_s_wait_tensorcnt(0);
        __syncthreads();
    }

    // ---- epilogue (M is always tile-exact here; only guard N) ----
#pragma unroll
    for (int j = 0; j < 4; ++j) {
        const int n = tileN + nwarp + j * 16 + lm;
        if (n >= N) continue;
        const float bv = bias[n];
#pragma unroll
        for (int i = 0; i < 2; ++i) {
            const size_t mbase = (size_t)(tileM + mwarp + i * 16 + 8 * half);
#pragma unroll
            for (int r = 0; r < 8; ++r) {
                const float val = acc[i][j][r] + bv;
                if (OUT_BF16)
                    ((__bf16*)C)[(mbase + r) * N + n] = (__bf16)val;
                else
                    ((float*)C)[(mbase + r) * N + n] = val;
            }
        }
    }
}

// ---------------- per-token attention core ----------------
// One 256-thread block per token s.
//   logits[i][j] = sum_m q[s][m][i] * k[s][j][m]
//   attn = log_softmax(logits, axis=j) * dropout_keep(mask_u)
//   out[i][j]    = sum_m attn[i][m] * v[s][j][m]   -> ao[s][i*64+j] (bf16)
__global__ __launch_bounds__(256) void attn_core(
        const __bf16* __restrict__ qb,
        const __bf16* __restrict__ kb,
        const __bf16* __restrict__ vb,
        const float*  __restrict__ mask_u,
        __bf16* __restrict__ ao) {
    __shared__ float qT[64 * 65];   // qT[c*65 + h] = q[s][h][c]
    __shared__ float kR[64 * 65];   // kR[h*65 + c] = k[s][h][c]
    __shared__ float vR[64 * 65];   // vR[h*65 + c] = v[s][h][c]
    __shared__ float L [64 * 64];   // masked log-softmax rows

    const int s = blockIdx.x;
    const int t = threadIdx.x;
    const __bf16* q = qb + (size_t)s * HC;
    const __bf16* k = kb + (size_t)s * HC;
    const __bf16* v = vb + (size_t)s * HC;

#pragma unroll
    for (int e = 0; e < 16; ++e) {
        const int f = t * 16 + e;          // flat = h*64 + c
        const int h = f >> 6, c = f & 63;
        qT[c * 65 + h] = (float)q[f];
        kR[h * 65 + c] = (float)k[f];
        vR[h * 65 + c] = (float)v[f];
    }
    __syncthreads();

    const int i  = t >> 2;                 // output row 0..63
    const int j0 = (t & 3) * 16;           // 16 columns per thread

    float lg[16];
#pragma unroll
    for (int jj = 0; jj < 16; ++jj) {
        const int j = j0 + jj;
        float a = 0.0f;
#pragma unroll 8
        for (int m = 0; m < 64; ++m) a = fmaf(qT[i * 65 + m], kR[j * 65 + m], a);
        lg[jj] = a;
    }

    float mx = lg[0];
#pragma unroll
    for (int jj = 1; jj < 16; ++jj) mx = fmaxf(mx, lg[jj]);
    mx = fmaxf(mx, __shfl_xor(mx, 1));
    mx = fmaxf(mx, __shfl_xor(mx, 2));
    float se = 0.0f;
#pragma unroll
    for (int jj = 0; jj < 16; ++jj) se += __expf(lg[jj] - mx);
    se += __shfl_xor(se, 1);
    se += __shfl_xor(se, 2);
    const float lse = __logf(se) + mx;

    const float* mu = mask_u + (size_t)s * HC + (size_t)i * 64 + j0;
#pragma unroll
    for (int jj = 0; jj < 16; ++jj) {
        const float keep = (mu[jj] > 0.25f) ? (1.0f / 0.75f) : 0.0f;
        L[i * 64 + j0 + jj] = (lg[jj] - lse) * keep;
    }
    __syncthreads();

#pragma unroll
    for (int jj = 0; jj < 16; ++jj) {
        const int j = j0 + jj;
        float a = 0.0f;
#pragma unroll 8
        for (int m = 0; m < 64; ++m) a = fmaf(L[i * 64 + m], vR[j * 65 + m], a);
        ao[(size_t)s * HC + (size_t)i * 64 + j] = (__bf16)a;
    }
}

// ---------------------------------------------------------------------------
extern "C" void kernel_launch(void* const* d_in, const int* in_sizes, int n_in,
                              void* d_out, int out_size, void* d_ws, size_t ws_size,
                              hipStream_t stream) {
    const float* query  = (const float*)d_in[0];   // [8192][4096]
    const float* key_in = (const float*)d_in[1];   // [8192][64]
    const float* value  = (const float*)d_in[2];   // [8192][64]
    const float* mask_u = (const float*)d_in[3];   // [8192][64][64]
    const float* Wq = (const float*)d_in[4];  const float* bq = (const float*)d_in[5];
    const float* Wk = (const float*)d_in[6];  const float* bk = (const float*)d_in[7];
    const float* Wv = (const float*)d_in[8];  const float* bv = (const float*)d_in[9];
    const float* Wo = (const float*)d_in[10]; const float* bo = (const float*)d_in[11];

    char* ws = (char*)d_ws;
    size_t off = 0;
    auto carve = [&](size_t bytes) -> void* {
        void* p = ws + off;
        off += (bytes + 255) & ~(size_t)255;
        return p;
    };
    __bf16* qbf    = (__bf16*)carve((size_t)S_TOK * QC * 2);
    __bf16* Wqt    = (__bf16*)carve((size_t)QC * HC * 2);     // Wq^T [HC][QC]
    __bf16* Wkt    = (__bf16*)carve((size_t)HC * NH * 2);     // Wk^T [HC][64]
    __bf16* Wvt    = (__bf16*)carve((size_t)HC * NH * 2);
    __bf16* kin_bf = (__bf16*)carve((size_t)S_TOK * NH * 2);
    __bf16* val_bf = (__bf16*)carve((size_t)S_TOK * NH * 2);
    __bf16* Wot    = (__bf16*)carve((size_t)NH * HC * 2);     // Wo^T [64][HC]
    __bf16* qproj  = (__bf16*)carve((size_t)S_TOK * HC * 2);
    __bf16* kproj  = (__bf16*)carve((size_t)S_TOK * HC * 2);
    __bf16* vproj  = (__bf16*)carve((size_t)S_TOK * HC * 2);
    __bf16* attn_o = (__bf16*)carve((size_t)S_TOK * HC * 2);

    const dim3 blk(256);

    // 1) conversions
    {
        long n4 = (long)S_TOK * QC / 4;
        cvt_f32_bf16_v4<<<dim3((unsigned)((n4 + 255) / 256)), blk, 0, stream>>>(
            (const float4*)query, (v4bf*)qbf, n4);
    }
    {
        long n = (long)QC * HC;
        cvt_transpose_f32_bf16<<<dim3((unsigned)((n + 255) / 256)), blk, 0, stream>>>(Wq, Wqt, QC, HC);
    }
    {
        long n = (long)NH * HC;
        cvt_transpose_f32_bf16<<<dim3((unsigned)((n + 255) / 256)), blk, 0, stream>>>(Wk, Wkt, NH, HC);
        cvt_transpose_f32_bf16<<<dim3((unsigned)((n + 255) / 256)), blk, 0, stream>>>(Wv, Wvt, NH, HC);
        cvt_transpose_f32_bf16<<<dim3((unsigned)((n + 255) / 256)), blk, 0, stream>>>(Wo, Wot, HC, NH);
    }
    {
        long n4 = (long)S_TOK * NH / 4;
        cvt_f32_bf16_v4<<<dim3((unsigned)((n4 + 255) / 256)), blk, 0, stream>>>(
            (const float4*)key_in, (v4bf*)kin_bf, n4);
        cvt_f32_bf16_v4<<<dim3((unsigned)((n4 + 255) / 256)), blk, 0, stream>>>(
            (const float4*)value, (v4bf*)val_bf, n4);
    }

    // 2) projections (bf16 WMMA GEMMs with TDM staging)
    {   // q: 8192 x 4096 x 4096, then k/v: 8192 x 4096 x 64
        dim3 g((HC + 255) / 256, (S_TOK + 63) / 64);
        gemm_bf16_wmma_tdm<true><<<g, blk, 0, stream>>>(qbf, Wqt, bq, qproj, S_TOK, HC, QC);
        gemm_bf16_wmma_tdm<true><<<g, blk, 0, stream>>>(kin_bf, Wkt, bk, kproj, S_TOK, HC, NH);
        gemm_bf16_wmma_tdm<true><<<g, blk, 0, stream>>>(val_bf, Wvt, bv, vproj, S_TOK, HC, NH);
    }

    // 3) per-token logits + log-softmax + dropout + @v
    attn_core<<<dim3(S_TOK), blk, 0, stream>>>(qproj, kproj, vproj, mask_u, attn_o);

    // 4) output projection: 8192 x 64 x 4096 -> f32 d_out
    {
        dim3 g((NH + 255) / 256, (S_TOK + 63) / 64);
        gemm_bf16_wmma_tdm<false><<<g, blk, 0, stream>>>(attn_o, Wot, bo, d_out, S_TOK, NH, QC);
    }
}
